// NNTransformLearner_21741124452436
// MI455X (gfx1250) — compile-verified, matrix operand order checked by reading, not run on hardware
//
#include <hip/hip_runtime.h>
#include <hip/hip_bf16.h>

// ---------------------------------------------------------------------------
// Gaussian-kernel attention (flash-attention style) for MI455X / gfx1250.
//   Q = X[:-1], K = y@R.T + t, V = X[1:]
//   logit l_ij = min(2g*q.k - g||q||^2 - g||k||^2, 0),  g = 1/256
//   out = (exp(l) @ V) / rowsum(exp(l))
// Compute-bound (~34 GFLOP vs ~13 MB traffic) -> bf16 WMMA, f32 accumulate.
// Split-K=4 over the key dimension (512 WGs = 2048 waves, ~2 per SIMD32) with
// f32 global atomics to combine partials; double-buffered
// GLOBAL_LOAD_ASYNC_TO_LDS_B128 streams the next 32-key tile during compute.
// ---------------------------------------------------------------------------

typedef __attribute__((ext_vector_type(16))) __bf16 v16bf;
typedef __attribute__((ext_vector_type(8)))  float  v8f;

#define GAMMAK (1.0f / 256.0f)
#define NPAD   8192          // 8191 real rows padded to 8192
#define D      128
#define SPLITK 4             // key-dimension splits per query block
#define NT_ALL (NPAD / 32)   // 256 key tiles total
#define NT_SP  (NT_ALL / SPLITK)

// round-to-nearest-even f32 -> bf16 (as raw u16)
__device__ __forceinline__ unsigned short f2bf(float f) {
  unsigned u = __float_as_uint(f);
  u += 0x7fffu + ((u >> 16) & 1u);
  return (unsigned short)(u >> 16);
}

// Low 32 bits of a generic pointer into __shared__ == LDS byte address.
__device__ __forceinline__ unsigned lds_addr(const void* p) {
  return (unsigned)(uintptr_t)p;
}

// Load one 16x32 bf16 WMMA fragment half-row for this lane:
// elements [0..7] at p, elements [8..15] at p+16 (per CDNA5 A/B VGPR layout).
__device__ __forceinline__ v16bf ld_frag(const unsigned short* p) {
  union { uint4 q[2]; v16bf v; } u;
  u.q[0] = *reinterpret_cast<const uint4*>(p);
  u.q[1] = *reinterpret_cast<const uint4*>(p + 16);
  return u.v;
}

// Async global -> LDS, 16B per lane, GVS mode (SGPR64 base + i32 VGPR offset).
__device__ __forceinline__ void async_b128(unsigned lds, int goff,
                                           const void* base) {
  asm volatile("global_load_async_to_lds_b128 %0, %1, %2 offset:0"
               :
               : "v"(lds), "v"(goff), "s"((unsigned long long)(uintptr_t)base)
               : "memory");
}

__device__ __forceinline__ void wait_async0() {
  asm volatile("s_wait_asynccnt 0x0" ::: "memory");
}

// ---------------------------------------------------------------------------
// Kernel 1: K-side prep.  Kb[i][:] = bf16(y_i @ R.T + t), kkb[i] = g*||k_i||^2
// Row 8191 is padding: Kb = 0, kkb = 1e30 (=> exp weight 0).
// ---------------------------------------------------------------------------
__global__ __launch_bounds__(128) void k_prep_k(
    const float* __restrict__ y, const float* __restrict__ R,
    const float* __restrict__ t, unsigned short* __restrict__ Kb,
    float* __restrict__ kkb) {
  const int i = blockIdx.x, j = threadIdx.x;
  __shared__ float red[128];
  float acc = 0.f;
  if (i < NPAD - 1) {
    acc = t[j];
    const float* yr = y + i * D;
    const float* Rr = R + j * D;           // (R.T)[k][j] == R[j][k]
#pragma unroll 8
    for (int k = 0; k < D; ++k) acc = fmaf(yr[k], Rr[k], acc);
  }
  Kb[i * D + j] = f2bf(acc);
  red[j] = acc * acc;
  __syncthreads();
  for (int s = 64; s > 0; s >>= 1) {
    if (j < s) red[j] += red[j + s];
    __syncthreads();
  }
  if (j == 0) kkb[i] = (i < NPAD - 1) ? GAMMAK * red[0] : 1.0e30f;
}

// ---------------------------------------------------------------------------
// Kernel 2: Q/V prep.  Qb[i][:] = bf16(X_i), qqb[i] = g*||q_i||^2,
// Vt[c][j] = bf16(X[j+1][c])  (transposed so PV B-fragments load contiguously)
// ---------------------------------------------------------------------------
__global__ __launch_bounds__(128) void k_prep_q(
    const float* __restrict__ X, unsigned short* __restrict__ Qb,
    unsigned short* __restrict__ Vt, float* __restrict__ qqb) {
  const int i = blockIdx.x, c = threadIdx.x;
  __shared__ float red[128];
  const float xq = (i < NPAD - 1) ? X[i * D + c] : 0.f;
  const float xv = (i < NPAD - 1) ? X[(i + 1) * D + c] : 0.f;
  Qb[i * D + c] = f2bf(xq);
  Vt[c * NPAD + i] = f2bf(xv);
  red[c] = xq * xq;
  __syncthreads();
  for (int s = 64; s > 0; s >>= 1) {
    if (c < s) red[c] += red[c + s];
    __syncthreads();
  }
  if (c == 0) qqb[i] = GAMMAK * red[0];
}

// ---------------------------------------------------------------------------
// Kernel 3: flash attention, split-K partials.
// Block = 128 threads = 4 waves; each wave owns 16 query rows (64 per block).
// blockIdx = query-block * SPLITK + split; each WG covers 64 key tiles.
// Partials accumulated with hardware global_atomic_add_f32.
// ---------------------------------------------------------------------------
__global__ __launch_bounds__(128) void k_flash(
    const unsigned short* __restrict__ Qb, const unsigned short* __restrict__ Kb,
    const unsigned short* __restrict__ Vt, const float* __restrict__ qqb,
    const float* __restrict__ kkb, float* __restrict__ num,
    float* __restrict__ den) {
  __shared__ alignas(16) unsigned short sK[2][32 * D];    // 32 keys x 128 ch
  __shared__ alignas(16) unsigned short sV[2][D * 32];    // 128 ch  x 32 keys
  __shared__ alignas(16) unsigned short sP[4 * 16 * 32];  // per-wave P tiles

  const int tid = threadIdx.x;
  const int wave = tid >> 5, lane = tid & 31;
  const int r = lane & 15, h = lane >> 4;
  const int qb = blockIdx.x >> 2, sp = blockIdx.x & (SPLITK - 1);
  const int q0 = qb * 64 + wave * 16;
  const int it0 = sp * NT_SP;            // first key tile of this split

  // Q fragments live in VGPRs for the whole key loop (4 x v16bf = 32 VGPRs).
  v16bf qf[4];
#pragma unroll
  for (int ci = 0; ci < 4; ++ci)
    qf[ci] = ld_frag(Qb + (q0 + r) * D + ci * 32 + 8 * h);

  float qbias[8];
#pragma unroll
  for (int v = 0; v < 8; ++v) qbias[v] = qqb[q0 + v + 8 * h];

  const v8f zero8 = {0.f, 0.f, 0.f, 0.f, 0.f, 0.f, 0.f, 0.f};
  v8f acc[8];
#pragma unroll
  for (int g = 0; g < 8; ++g) acc[g] = zero8;
  v8f accD = zero8;                       // denominator via ones-WMMA

  union { uint4 q[2]; v16bf v; } onesu;
  onesu.q[0] = make_uint4(0x3F803F80u, 0x3F803F80u, 0x3F803F80u, 0x3F803F80u);
  onesu.q[1] = onesu.q[0];
  const v16bf ones = onesu.v;

  unsigned short* sPw = sP + wave * (16 * 32);

  // Issue async tile loads for key block j0 into buffer b.
  auto issue_tile = [&](int j0, int b) {
#pragma unroll
    for (int k = 0; k < 4; ++k) {
      const int idx = tid + k * 128;                 // 512 x 16B per tile
      // K tile: rows j0..j0+31 are contiguous (32*128 bf16 = 8 KB)
      async_b128(lds_addr(&sK[b][idx * 8]), j0 * (D * 2) + idx * 16, Kb);
      // Vt tile: 128 rows of 64B each
      const int row = idx >> 2, c4 = idx & 3;
      async_b128(lds_addr(&sV[b][idx * 8]),
                 row * (NPAD * 2) + j0 * 2 + c4 * 16, Vt);
    }
  };

  issue_tile(it0 * 32, 0);  // prologue

  for (int n = 0; n < NT_SP; ++n) {
    const int b = n & 1;
    const int j0 = (it0 + n) * 32;

    wait_async0();     // my async writes for this tile have landed in LDS
    __syncthreads();   // everyone's writes visible; prev reads of buf 1-b done

    if (n + 1 < NT_SP) issue_tile(j0 + 32, 1 - b);  // overlaps with compute

    // S = Q K^T over two 16-key subtiles; exp -> bf16 P staged in LDS
#pragma unroll
    for (int jt = 0; jt < 2; ++jt) {
      // Preload all four B-fragments, then drain into WMMAs.
      v16bf kf[4];
#pragma unroll
      for (int ci = 0; ci < 4; ++ci)
        kf[ci] = ld_frag(&sK[b][(jt * 16 + r) * D + ci * 32 + 8 * h]);
      v8f s = zero8;
#pragma unroll
      for (int ci = 0; ci < 4; ++ci)
        s = __builtin_amdgcn_wmma_f32_16x16x32_bf16(false, qf[ci], false, kf[ci],
                                                    (short)0, s, false, false);
      const float kbias = kkb[j0 + jt * 16 + r];
#pragma unroll
      for (int v = 0; v < 8; ++v) {
        const float l = s[v] * (2.0f * GAMMAK) - qbias[v] - kbias;
        const float p = __expf(fminf(l, 0.0f));   // clip(sq,0) => l <= 0
        sPw[(v + 8 * h) * 32 + jt * 16 + r] = f2bf(p);  // D-layout -> A-layout
      }
    }

    // P fragment (16x32) back from LDS; denominator + PV accumulation
    v16bf pf = ld_frag(&sPw[r * 32 + 8 * h]);
    accD = __builtin_amdgcn_wmma_f32_16x16x32_bf16(false, pf, false, ones,
                                                   (short)0, accD, false, false);
#pragma unroll
    for (int gp = 0; gp < 8; gp += 4) {
      v16bf vf[4];
#pragma unroll
      for (int u = 0; u < 4; ++u)
        vf[u] = ld_frag(&sV[b][((gp + u) * 16 + r) * 32 + 8 * h]);
#pragma unroll
      for (int u = 0; u < 4; ++u)
        acc[gp + u] = __builtin_amdgcn_wmma_f32_16x16x32_bf16(
            false, pf, false, vf[u], (short)0, acc[gp + u], false, false);
    }
  }

  // Epilogue: combine split-K partials with hardware f32 atomics.
#pragma unroll
  for (int v = 0; v < 8; ++v) {
    const int q = q0 + v + 8 * h;
    if (q < NPAD - 1) {
      if (r == 0) unsafeAtomicAdd(&den[q], accD[v]);  // all 16 cols identical
#pragma unroll
      for (int g = 0; g < 8; ++g)
        unsafeAtomicAdd(&num[q * D + g * 16 + r], acc[g][v]);
    }
  }
}

// ---------------------------------------------------------------------------
// Kernel 4: out = num / den
// ---------------------------------------------------------------------------
__global__ __launch_bounds__(128) void k_combine(
    const float* __restrict__ num, const float* __restrict__ den,
    float* __restrict__ out) {
  const int i = blockIdx.x, c = threadIdx.x;
  out[i * D + c] = num[i * D + c] / den[i];
}

// ---------------------------------------------------------------------------
extern "C" void kernel_launch(void* const* d_in, const int* in_sizes, int n_in,
                              void* d_out, int out_size, void* d_ws, size_t ws_size,
                              hipStream_t stream) {
  const float* X = (const float*)d_in[0];   // (8192,128)
  const float* y = (const float*)d_in[1];   // (8191,128)
  // d_in[2] = y_next (unused by reference)
  const float* R = (const float*)d_in[3];   // (128,128)
  const float* t = (const float*)d_in[4];   // (128,1)
  float* out = (float*)d_out;               // (8191,128)

  // Workspace (~10.4 MB): Qb | Kb | Vt (bf16) | qqb | kkb | num | den (f32)
  unsigned short* Qb = (unsigned short*)d_ws;
  unsigned short* Kb = Qb + (size_t)NPAD * D;
  unsigned short* Vt = Kb + (size_t)NPAD * D;
  float* qqb = (float*)(Vt + (size_t)D * NPAD);
  float* kkb = qqb + NPAD;
  float* num = kkb + NPAD;                  // NPAD*D floats
  float* den = num + (size_t)NPAD * D;      // NPAD floats

  k_prep_k<<<NPAD, 128, 0, stream>>>(y, R, t, Kb, kkb);
  k_prep_q<<<NPAD, 128, 0, stream>>>(X, Qb, Vt, qqb);
  hipMemsetAsync(num, 0, ((size_t)NPAD * D + NPAD) * sizeof(float), stream);
  k_flash<<<(NPAD / 64) * SPLITK, 128, 0, stream>>>(Qb, Kb, Vt, qqb, kkb, num, den);
  k_combine<<<NPAD - 1, 128, 0, stream>>>(num, den, out);
}
